// GraphPlayerEncoder_14448269984517
// MI455X (gfx1250) — compile-verified, matrix operand order checked by reading, not run on hardware
//
#include <hip/hip_runtime.h>

#define N_NODES 20000
#define N_EDGES 160000
#define EA_TOT  (N_EDGES + N_NODES)   // 180000 (edges + self loops)
#define NB      1000
#define D_IN    7
#define HID     64
#define HEADS   4
#define CTXD    3
#define EDIM    5
#define L_LAYERS 4
#define PROJC   (HEADS * HID)         // 256

typedef __attribute__((ext_vector_type(16))) _Float16 v16h;
typedef __attribute__((ext_vector_type(8)))  float    v8f;

// ---------------- WMMA helpers (CDNA5 wave32, 16x16x32 f16 -> f32) -------------
// Fragment loader for a 16-row, K-contiguous LDS tile (ld in halves, 16B-aligned
// rows). Works for both the A matrix (row = lane&15) and a transposed-B tile
// (col = lane&15): halves 0..7 -> K=kb+0..7, halves 8..15 -> K=kb+16..23 with
// kb=(lane>>4)*8. All 8-half runs are contiguous -> lowers to ds_load_b128.
__device__ __forceinline__ v16h load_frag(const _Float16* P, int ld, int lane) {
  v16h f;
  const int m  = lane & 15;
  const int kb = (lane >> 4) << 3;
#pragma unroll
  for (int i = 0; i < 16; ++i) {
    const int k = kb + i + ((i >= 8) ? 8 : 0);
    f[i] = P[m * ld + k];
  }
  return f;
}
__device__ __forceinline__ v8f wmma_f(v16h a, v16h b, v8f c) {
  return __builtin_amdgcn_wmma_f32_16x16x32_f16(false, a, false, b, (short)0, c, false, false);
}

__device__ __forceinline__ float wave_reduce_add(float v) {
#pragma unroll
  for (int off = 16; off > 0; off >>= 1) v += __shfl_xor(v, off, 32);
  return v;
}

// ordered-uint encoding for float atomicMax (init 0 == smaller than any real value)
__device__ __forceinline__ unsigned int f2enc(float f) {
  unsigned int u = __float_as_uint(f);
  return (u & 0x80000000u) ? ~u : (u | 0x80000000u);
}
__device__ __forceinline__ float enc2f(unsigned int e) {
  unsigned int u = (e & 0x80000000u) ? (e & 0x7FFFFFFFu) : ~e;
  return __uint_as_float(u);
}

// ------------------------------- utility ---------------------------------------
__global__ void fill_kernel(float* __restrict__ p, long n, float v) {
  long i = (long)blockIdx.x * blockDim.x + threadIdx.x;
  long stride = (long)gridDim.x * blockDim.x;
  for (; i < n; i += stride) p[i] = v;
}

// ------------------------------ front end --------------------------------------
__global__ void ctx_kernel(const float* __restrict__ context, const int* __restrict__ formation,
                           const int* __restrict__ alignment,
                           const float* __restrict__ ctx_W, const float* __restrict__ ctx_b,
                           const float* __restrict__ form_tab, const float* __restrict__ align_tab,
                           float* __restrict__ c_ctx) {
  int bb = blockIdx.x, c = threadIdx.x;
  float acc = ctx_b[c];
#pragma unroll
  for (int k = 0; k < CTXD; ++k) acc += context[bb * CTXD + k] * ctx_W[k * HID + c];
  acc = fmaxf(acc, 0.f);
  acc += form_tab[formation[bb] * HID + c] + align_tab[alignment[bb] * HID + c];
  c_ctx[(long)bb * HID + c] = acc;
}

__global__ void embed_kernel(const float* __restrict__ x, const int* __restrict__ role,
                             const int* __restrict__ side, const int* __restrict__ batch,
                             const float* __restrict__ frame_t,
                             const float* __restrict__ emb_W, const float* __restrict__ emb_b,
                             const float* __restrict__ role_tab, const float* __restrict__ side_tab,
                             const float* __restrict__ temp_tab,
                             const float* __restrict__ c_ctx, float* __restrict__ H) {
  int n = blockIdx.x, c = threadIdx.x;
  float acc = emb_b[c];
#pragma unroll
  for (int k = 0; k < D_IN; ++k) acc += x[(long)n * D_IN + k] * emb_W[k * HID + c];
  acc = fmaxf(acc, 0.f);
  acc += role_tab[role[n] * HID + c];
  if (c < HID / 2) acc += side_tab[side[n] * (HID / 2) + c];
  int tix = (int)(frame_t[0] * 99.0f);
  tix = min(max(tix, 0), 99);
  acc += temp_tab[tix * HID + c];
  acc += c_ctx[(long)batch[n] * HID + c];
  H[(long)n * HID + c] = acc;
}

__global__ void prep_edges_kernel(const int* __restrict__ src, const int* __restrict__ dst,
                                  const float* __restrict__ edge_attr,
                                  float* __restrict__ in_cnt, float* __restrict__ loop_sum,
                                  float* __restrict__ cnt_src) {
  int e = blockIdx.x * 256 + threadIdx.x;
  if (e >= N_EDGES) return;
  int d = dst[e], s = src[e];
  atomicAdd(&in_cnt[d], 1.0f);
  atomicAdd(&cnt_src[s], 1.0f);
#pragma unroll
  for (int k = 0; k < EDIM; ++k)
    atomicAdd(&loop_sum[(long)d * EDIM + k], edge_attr[(long)e * EDIM + k]);
}

// -------------- WMMA projection: [32 rows,64] @ [64,256] + bias ----------------
__global__ __launch_bounds__(128)
void proj_kernel(const float* __restrict__ A, const float* __restrict__ W,
                 const float* __restrict__ bias, float* __restrict__ Out) {
  __shared__ _Float16 As[32][72];      // 32 rows x K=64
  __shared__ _Float16 sWt[64][72];     // transposed: [col][K]
  const int tid = threadIdx.x;
  const int row0 = blockIdx.x * 32;
  const int col0 = blockIdx.y * 64;
  for (int i = tid; i < 32 * 16; i += 128) {            // A: float4 loads
    int r = i >> 4, c4 = (i & 15) << 2;
    const float4 v = *(const float4*)&A[(long)(row0 + r) * HID + c4];
    As[r][c4 + 0] = (_Float16)v.x; As[r][c4 + 1] = (_Float16)v.y;
    As[r][c4 + 2] = (_Float16)v.z; As[r][c4 + 3] = (_Float16)v.w;
  }
  for (int i = tid; i < 64 * 16; i += 128) {            // W: float4 loads, transposed store
    int k = i >> 4, c4 = (i & 15) << 2;
    const float4 v = *(const float4*)&W[(long)k * PROJC + col0 + c4];
    sWt[c4 + 0][k] = (_Float16)v.x; sWt[c4 + 1][k] = (_Float16)v.y;
    sWt[c4 + 2][k] = (_Float16)v.z; sWt[c4 + 3][k] = (_Float16)v.w;
  }
  __syncthreads();
  const int wave = tid >> 5, lane = tid & 31;
  const int n0 = wave * 16;
  const int nn = lane & 15, mrow = (lane >> 4) * 8;
  const float bv = bias[col0 + n0 + nn];
#pragma unroll
  for (int rt = 0; rt < 2; ++rt) {
    v8f acc = {};
#pragma unroll
    for (int ks = 0; ks < 2; ++ks) {
      v16h a  = load_frag(&As[rt * 16][ks * 32], 72, lane);
      v16h bf = load_frag(&sWt[n0][ks * 32], 72, lane);
      acc = wmma_f(a, bf, acc);
    }
#pragma unroll
    for (int r = 0; r < 8; ++r)
      Out[(long)(row0 + rt * 16 + mrow + r) * PROJC + col0 + n0 + nn] = acc[r] + bv;
  }
}

// --------------------------- GATv2 edge attention -------------------------------
__global__ __launch_bounds__(256)
void gat_alpha_kernel(const float* __restrict__ xl, const float* __restrict__ xr,
                      const int* __restrict__ src, const int* __restrict__ dst,
                      const float* __restrict__ edge_attr,
                      const float* __restrict__ loop_sum, const float* __restrict__ in_cnt,
                      const float* __restrict__ We, const float* __restrict__ att,
                      float* __restrict__ alpha, unsigned int* __restrict__ mmax) {
  __shared__ float sWe[EDIM * PROJC];
  __shared__ float sAtt[PROJC];
  __shared__ float sEA[4][EDIM];
  __shared__ int   sSrc[4], sDst[4];
  __shared__ float red2[8][HEADS];    // per-wave partials (2 waves per edge)
  const int tid = threadIdx.x;
  const int el = tid >> 6, c = tid & 63;
  const long e = (long)blockIdx.x * 4 + el;
  for (int i = tid; i < EDIM * PROJC; i += 256) sWe[i] = We[i];
  sAtt[tid] = att[tid];
  if (c == 0) {
    int s, d;
    if (e < N_EDGES) { s = src[e]; d = dst[e]; } else { s = d = (int)(e - N_EDGES); }
    sSrc[el] = s; sDst[el] = d;
  }
  if (c < EDIM) {
    float v;
    if (e < N_EDGES) v = edge_attr[e * EDIM + c];
    else { int n = (int)(e - N_EDGES); v = loop_sum[(long)n * EDIM + c] / fmaxf(in_cnt[n], 1.0f); }
    sEA[el][c] = v;
  }
  __syncthreads();
  const int s = sSrc[el], d = sDst[el];
#pragma unroll
  for (int h = 0; h < HEADS; ++h) {
    float ev = 0.f;
#pragma unroll
    for (int k = 0; k < EDIM; ++k) ev += sEA[el][k] * sWe[k * PROJC + h * HID + c];
    float m = xl[(long)s * PROJC + h * HID + c] + xr[(long)d * PROJC + h * HID + c] + ev;
    m = (m > 0.f) ? m : 0.2f * m;     // leaky_relu(0.2)
    float v = wave_reduce_add(m * sAtt[h * HID + c]);
    if ((tid & 31) == 0) red2[tid >> 5][h] = v;
  }
  __syncthreads();
  if (c < HEADS) {
    float a = red2[el * 2][c] + red2[el * 2 + 1][c];
    alpha[e * HEADS + c] = a;
    atomicMax(&mmax[(long)d * HEADS + c], f2enc(a));
  }
}

__global__ void gat_expsum_kernel(const int* __restrict__ dst, float* __restrict__ alpha,
                                  const unsigned int* __restrict__ mmax, float* __restrict__ ssum) {
  long i = (long)blockIdx.x * blockDim.x + threadIdx.x;
  if (i >= (long)EA_TOT * HEADS) return;
  long e = i >> 2; int h = (int)(i & 3);
  int d = (e < N_EDGES) ? dst[e] : (int)(e - N_EDGES);
  float m = enc2f(mmax[(long)d * HEADS + h]);
  float ex = __expf(alpha[i] - m);
  alpha[i] = ex;
  atomicAdd(&ssum[(long)d * HEADS + h], ex);
}

__global__ __launch_bounds__(256)
void gat_agg_kernel(const float* __restrict__ xl, const int* __restrict__ src,
                    const int* __restrict__ dst, const float* __restrict__ alpha,
                    const float* __restrict__ ssum, float* __restrict__ agg) {
  __shared__ float sw[4][HEADS];
  __shared__ int sSrc[4], sDst[4];
  const int tid = threadIdx.x;
  const int el = tid >> 6, c = tid & 63;
  const long e = (long)blockIdx.x * 4 + el;
  if (c == 0) {
    int s, d;
    if (e < N_EDGES) { s = src[e]; d = dst[e]; } else { s = d = (int)(e - N_EDGES); }
    sSrc[el] = s; sDst[el] = d;
  }
  __syncthreads();
  if (c < HEADS) {
    int d = sDst[el];
    sw[el][c] = alpha[e * HEADS + c] / (ssum[(long)d * HEADS + c] + 1e-16f);
  }
  __syncthreads();
  const int s = sSrc[el], d = sDst[el];
  float v = 0.f;
#pragma unroll
  for (int h = 0; h < HEADS; ++h) v += sw[el][h] * xl[(long)s * PROJC + h * HID + c];
  atomicAdd(&agg[(long)d * HID + c], 0.25f * v);   // mean over heads folded in
}

// relu(agg + bias) + residual, then LayerNorm -> H (in place per node)
__global__ __launch_bounds__(64)
void gat_post_kernel(const float* __restrict__ agg, float* __restrict__ H,
                     const float* __restrict__ bias, const float* __restrict__ g,
                     const float* __restrict__ b) {
  __shared__ float s2[2];
  const int n = blockIdx.x, c = threadIdx.x;
  float hn = fmaxf(agg[(long)n * HID + c] + bias[c], 0.f);
  float t = hn + H[(long)n * HID + c];
  float vs = wave_reduce_add(t);
  if ((c & 31) == 0) s2[c >> 5] = vs;
  __syncthreads();
  float mu = (s2[0] + s2[1]) / 64.f;
  __syncthreads();
  float dd = t - mu;
  float vq = wave_reduce_add(dd * dd);
  if ((c & 31) == 0) s2[c >> 5] = vq;
  __syncthreads();
  float var = (s2[0] + s2[1]) / 64.f;
  H[(long)n * HID + c] = dd * rsqrtf(var + 1e-5f) * g[c] + b[c];
}

// ------- fused social pooling: 3 chained WMMA GEMMs per 32-edge tile ------------
__global__ __launch_bounds__(128)
void social_kernel(const float* __restrict__ H, const int* __restrict__ src,
                   const int* __restrict__ dst,
                   const float* __restrict__ W1, const float* __restrict__ b1,
                   const float* __restrict__ W2, const float* __restrict__ b2,
                   const float* __restrict__ Wg, const float* __restrict__ bg,
                   float* __restrict__ pooled) {
  __shared__ _Float16 sPair[32][136];   // 32 x 128
  __shared__ _Float16 sW1t[64][136];    // transposed: [col][K=128]
  __shared__ _Float16 sW2t[64][72];     // transposed: [col][K=64]
  __shared__ _Float16 sWgt[64][72];
  __shared__ _Float16 sT1[32][72];
  __shared__ _Float16 sI[32][72];
  const int tid = threadIdx.x;
  const long e0 = (long)blockIdx.x * 32;
  for (int i = tid; i < 128 * 16; i += 128) {            // W1 [128,64] transposed
    int k = i >> 4, c4 = (i & 15) << 2;
    const float4 v = *(const float4*)&W1[(long)k * 64 + c4];
    sW1t[c4 + 0][k] = (_Float16)v.x; sW1t[c4 + 1][k] = (_Float16)v.y;
    sW1t[c4 + 2][k] = (_Float16)v.z; sW1t[c4 + 3][k] = (_Float16)v.w;
  }
  for (int i = tid; i < 64 * 16; i += 128) {             // W2 [64,64] transposed
    int k = i >> 4, c4 = (i & 15) << 2;
    const float4 v = *(const float4*)&W2[(long)k * 64 + c4];
    sW2t[c4 + 0][k] = (_Float16)v.x; sW2t[c4 + 1][k] = (_Float16)v.y;
    sW2t[c4 + 2][k] = (_Float16)v.z; sW2t[c4 + 3][k] = (_Float16)v.w;
  }
  for (int i = tid; i < 64 * 16; i += 128) {             // Wg [64,64] transposed
    int k = i >> 4, c4 = (i & 15) << 2;
    const float4 v = *(const float4*)&Wg[(long)k * 64 + c4];
    sWgt[c4 + 0][k] = (_Float16)v.x; sWgt[c4 + 1][k] = (_Float16)v.y;
    sWgt[c4 + 2][k] = (_Float16)v.z; sWgt[c4 + 3][k] = (_Float16)v.w;
  }
  for (int i = tid; i < 32 * 32; i += 128) {             // pair = [h[src] | h[dst]]
    int r = i >> 5, q = (i & 31) << 2;
    int node = (q < 64) ? src[e0 + r] : dst[e0 + r];
    const float4 v = *(const float4*)&H[(long)node * HID + (q & 63)];
    sPair[r][q + 0] = (_Float16)v.x; sPair[r][q + 1] = (_Float16)v.y;
    sPair[r][q + 2] = (_Float16)v.z; sPair[r][q + 3] = (_Float16)v.w;
  }
  __syncthreads();
  const int wave = tid >> 5, lane = tid & 31;
  const int n0 = wave * 16;
  const int nn = lane & 15, mrow = (lane >> 4) * 8;
  // stage 1: t1 = relu(pair @ W1 + b1)  (K=128 -> 4 WMMA per row-tile)
  {
    const float bv = b1[n0 + nn];
#pragma unroll
    for (int rt = 0; rt < 2; ++rt) {
      v8f acc = {};
#pragma unroll
      for (int ks = 0; ks < 4; ++ks) {
        v16h a  = load_frag(&sPair[rt * 16][ks * 32], 136, lane);
        v16h bf = load_frag(&sW1t[n0][ks * 32], 136, lane);
        acc = wmma_f(a, bf, acc);
      }
#pragma unroll
      for (int r = 0; r < 8; ++r)
        sT1[rt * 16 + mrow + r][n0 + nn] = (_Float16)fmaxf(acc[r] + bv, 0.f);
    }
  }
  __syncthreads();
  // stage 2: inter = t1 @ W2 + b2       (K=64 -> 2 WMMA per row-tile)
  v8f acc2[2];
  {
    const float bv = b2[n0 + nn];
#pragma unroll
    for (int rt = 0; rt < 2; ++rt) {
      v8f acc = {};
#pragma unroll
      for (int ks = 0; ks < 2; ++ks) {
        v16h a  = load_frag(&sT1[rt * 16][ks * 32], 72, lane);
        v16h bf = load_frag(&sW2t[n0][ks * 32], 72, lane);
        acc = wmma_f(a, bf, acc);
      }
#pragma unroll
      for (int r = 0; r < 8; ++r) {
        acc[r] = acc[r] + bv;
        sI[rt * 16 + mrow + r][n0 + nn] = (_Float16)acc[r];
      }
      acc2[rt] = acc;
    }
  }
  __syncthreads();
  // stage 3: g = inter @ Wg + bg; gated = inter * sigmoid(g); scatter-add by src
  {
    const float bv = bg[n0 + nn];
#pragma unroll
    for (int rt = 0; rt < 2; ++rt) {
      v8f acc = {};
#pragma unroll
      for (int ks = 0; ks < 2; ++ks) {
        v16h a  = load_frag(&sI[rt * 16][ks * 32], 72, lane);
        v16h bf = load_frag(&sWgt[n0][ks * 32], 72, lane);
        acc = wmma_f(a, bf, acc);
      }
#pragma unroll
      for (int r = 0; r < 8; ++r) {
        float gv = acc[r] + bv;
        float gated = acc2[rt][r] * (1.f / (1.f + __expf(-gv)));
        int row = rt * 16 + mrow + r;
        int s = src[e0 + row];
        atomicAdd(&pooled[(long)s * HID + n0 + nn], gated);
      }
    }
  }
}

// final: LN(h + pooled/max(cnt,1)) -> d_out
__global__ __launch_bounds__(64)
void final_ln_kernel(const float* __restrict__ H, const float* __restrict__ pooled,
                     const float* __restrict__ cnt_src, const float* __restrict__ g,
                     const float* __restrict__ b, float* __restrict__ out) {
  __shared__ float s2[2];
  const int n = blockIdx.x, c = threadIdx.x;
  float p = pooled[(long)n * HID + c] / fmaxf(cnt_src[n], 1.0f);
  float t = H[(long)n * HID + c] + p;
  float vs = wave_reduce_add(t);
  if ((c & 31) == 0) s2[c >> 5] = vs;
  __syncthreads();
  float mu = (s2[0] + s2[1]) / 64.f;
  __syncthreads();
  float dd = t - mu;
  float vq = wave_reduce_add(dd * dd);
  if ((c & 31) == 0) s2[c >> 5] = vq;
  __syncthreads();
  float var = (s2[0] + s2[1]) / 64.f;
  out[(long)n * HID + c] = dd * rsqrtf(var + 1e-5f) * g[c] + b[c];
}

// --------------------------------- launcher -------------------------------------
extern "C" void kernel_launch(void* const* d_in, const int* in_sizes, int n_in,
                              void* d_out, int out_size, void* d_ws, size_t ws_size,
                              hipStream_t stream) {
  const float* x         = (const float*)d_in[0];
  const int*   ei        = (const int*)d_in[1];
  const float* edge_attr = (const float*)d_in[2];
  const float* context   = (const float*)d_in[3];
  const int*   batch     = (const int*)d_in[4];
  const int*   role      = (const int*)d_in[5];
  const int*   side      = (const int*)d_in[6];
  const int*   formation = (const int*)d_in[7];
  const int*   alignment = (const int*)d_in[8];
  const float* frame_t   = (const float*)d_in[9];
  const float* emb_W     = (const float*)d_in[10];
  const float* emb_b     = (const float*)d_in[11];
  const float* role_tab  = (const float*)d_in[12];
  const float* side_tab  = (const float*)d_in[13];
  const float* ctx_W     = (const float*)d_in[14];
  const float* ctx_b     = (const float*)d_in[15];
  const float* form_tab  = (const float*)d_in[16];
  const float* align_tab = (const float*)d_in[17];
  const float* temp_tab  = (const float*)d_in[18];
  const float* gat_Wl    = (const float*)d_in[19];
  const float* gat_bl    = (const float*)d_in[20];
  const float* gat_Wr    = (const float*)d_in[21];
  const float* gat_br    = (const float*)d_in[22];
  const float* gat_We    = (const float*)d_in[23];
  const float* gat_att   = (const float*)d_in[24];
  const float* gat_bias  = (const float*)d_in[25];
  const float* ln_g      = (const float*)d_in[26];
  const float* ln_b      = (const float*)d_in[27];
  const float* sp_W1     = (const float*)d_in[28];
  const float* sp_b1     = (const float*)d_in[29];
  const float* sp_W2     = (const float*)d_in[30];
  const float* sp_b2     = (const float*)d_in[31];
  const float* sp_Wg     = (const float*)d_in[32];
  const float* sp_bg     = (const float*)d_in[33];
  const float* fn_g      = (const float*)d_in[34];
  const float* fn_b      = (const float*)d_in[35];

  const int* src  = ei;
  const int* dstp = ei + N_EDGES;

  float* ws = (float*)d_ws;
  long o = 0;
  float* c_ctx    = ws + o; o += (long)NB * HID;
  float* in_cnt   = ws + o; o += N_NODES;              // |
  float* loop_sum = ws + o; o += (long)N_NODES * EDIM; // | contiguous zero-once region
  float* cnt_src  = ws + o; o += N_NODES;              // |
  float* Hbuf     = ws + o; o += (long)N_NODES * HID;
  float* xl       = ws + o; o += (long)N_NODES * PROJC;
  float* xr       = ws + o; o += (long)N_NODES * PROJC;
  float* alpha    = ws + o; o += (long)EA_TOT * HEADS;
  float* agg      = ws + o; o += (long)N_NODES * HID;  // |
  unsigned int* mmax = (unsigned int*)(ws + o); o += (long)N_NODES * HEADS; // | zero-per-layer
  float* ssum     = ws + o; o += (long)N_NODES * HEADS;                     // |
  float* pooled   = ws + o; o += (long)N_NODES * HID;

  auto nb = [](long n) { return (int)((n + 255) / 256); };

  // zero-once: in_cnt+loop_sum+cnt_src, pooled
  const long zprep = (long)N_NODES * (2 + EDIM);
  fill_kernel<<<nb(zprep), 256, 0, stream>>>(in_cnt, zprep, 0.f);
  fill_kernel<<<nb((long)N_NODES * HID), 256, 0, stream>>>(pooled, (long)N_NODES * HID, 0.f);

  ctx_kernel<<<NB, HID, 0, stream>>>(context, formation, alignment, ctx_W, ctx_b,
                                     form_tab, align_tab, c_ctx);
  prep_edges_kernel<<<nb(N_EDGES), 256, 0, stream>>>(src, dstp, edge_attr,
                                                     in_cnt, loop_sum, cnt_src);
  embed_kernel<<<N_NODES, HID, 0, stream>>>(x, role, side, batch, frame_t, emb_W, emb_b,
                                            role_tab, side_tab, temp_tab, c_ctx, Hbuf);

  const long zlayer = (long)N_NODES * (HID + 2 * HEADS); // agg + mmax + ssum (contiguous)
  for (int i = 0; i < L_LAYERS; ++i) {
    const float* Wl = gat_Wl + (long)i * HID * PROJC;
    const float* bl = gat_bl + (long)i * PROJC;
    const float* Wr = gat_Wr + (long)i * HID * PROJC;
    const float* br = gat_br + (long)i * PROJC;
    const float* We = gat_We + (long)i * EDIM * PROJC;
    const float* at = gat_att + (long)i * HEADS * HID;
    const float* gb = gat_bias + (long)i * HID;
    const float* lg = ln_g + (long)i * HID;
    const float* lb = ln_b + (long)i * HID;

    fill_kernel<<<nb(zlayer), 256, 0, stream>>>(agg, zlayer, 0.f);

    dim3 pgrid(N_NODES / 32, PROJC / 64);   // 625 x 4, exact tiles
    proj_kernel<<<pgrid, 128, 0, stream>>>(Hbuf, Wl, bl, xl);
    proj_kernel<<<pgrid, 128, 0, stream>>>(Hbuf, Wr, br, xr);

    gat_alpha_kernel<<<EA_TOT / 4, 256, 0, stream>>>(xl, xr, src, dstp, edge_attr,
                                                     loop_sum, in_cnt, We, at, alpha, mmax);
    gat_expsum_kernel<<<nb((long)EA_TOT * HEADS), 256, 0, stream>>>(dstp, alpha, mmax, ssum);
    gat_agg_kernel<<<EA_TOT / 4, 256, 0, stream>>>(xl, src, dstp, alpha, ssum, agg);
    gat_post_kernel<<<N_NODES, HID, 0, stream>>>(agg, Hbuf, gb, lg, lb);
  }

  social_kernel<<<N_EDGES / 32, 128, 0, stream>>>(Hbuf, src, dstp, sp_W1, sp_b1, sp_W2,
                                                  sp_b2, sp_Wg, sp_bg, pooled);
  final_ln_kernel<<<N_NODES, HID, 0, stream>>>(Hbuf, pooled, cnt_src, fn_g, fn_b,
                                               (float*)d_out);
}
